// EventDrivenAttention_50964081934585
// MI455X (gfx1250) — matrix operand order repaired and checked by reading, not compiled, not run on hardware
//
#include <hip/hip_runtime.h>
#include <hip/hip_bf16.h>

typedef __bf16 bf16;
typedef __attribute__((ext_vector_type(8)))  float v8f;
typedef __attribute__((ext_vector_type(16))) bf16  v16bf;
typedef __attribute__((ext_vector_type(8)))  bf16  v8bf;
typedef __attribute__((ext_vector_type(4)))  unsigned int u32x4;
typedef __attribute__((ext_vector_type(4)))  int i32x4;
typedef __attribute__((ext_vector_type(8)))  int i32x8;

union FragBf {
    v16bf v;
    struct { v8bf lo; v8bf hi; } h;
};

static constexpr int cB  = 2;
static constexpr int cS  = 2048;
static constexpr int cH  = 1024;
static constexpr int cNH = 16;
static constexpr int cHD = 64;
static constexpr int cBS = cB * cS;        // 4096 rows
static constexpr float cScale = 0.125f;    // 1/sqrt(64)

__device__ __forceinline__ v8f wmma_bf16(v16bf a, v16bf b, v8f c) {
    return __builtin_amdgcn_wmma_f32_16x16x32_bf16(false, a, false, b, (short)0, c, false, false);
}

// uniform (SGPR) wave id -> guarantees a scalar branch around TDM issue,
// since tensor ops execute regardless of EXEC (ISA 08_async_tensor Sec 7.1).
__device__ __forceinline__ int wave_id_uniform() {
    return __builtin_amdgcn_readfirstlane((int)threadIdx.x) >> 5;
}

// ---------------------------------------------------------------------------
// TDM: DMA a 128-row x tile_w-col bf16 tile (row stride = stride_elems) from
// global into LDS at lds_addr, adding 16B padding after each row:
//   tile_w=32, pad_code=3 (pad after 16 DW = 64B)  -> [128][40] pitch
//   tile_w=64, pad_code=4 (pad after 32 DW = 128B) -> [128][72] pitch
// D# bitfields per CDNA5 ISA Sec 8.3 / 8.4.
__device__ __forceinline__ void tdm_load_tile_bf16(unsigned lds_addr,
                                                   const void* gptr,
                                                   unsigned stride_elems,
                                                   unsigned tile_w,
                                                   unsigned pad_code) {
    unsigned long long ga = (unsigned long long)gptr;
    u32x4 g0;
    g0[0] = 1u;                                    // count=1 (valid), user mode
    g0[1] = lds_addr;                              // LDS byte address
    g0[2] = (unsigned)ga;                          // global_addr[31:0]
    g0[3] = (unsigned)(ga >> 32) | (2u << 30);     // global_addr[56:32] | type=2
    i32x8 g1;
    g1[0] = (int)((1u << 16) | (1u << 20) | (pad_code << 22) | (3u << 25));
    g1[1] = (int)(tile_w << 16);                   // tensor_dim0[15:0] @ [63:48]
    g1[2] = (int)(128u << 16);                     // tensor_dim0 hi=0 | tensor_dim1 lo=128
    g1[3] = (int)(tile_w << 16);                   // tensor_dim1 hi=0 | tile_dim0
    g1[4] = (int)128u;                             // tile_dim1=128 | tile_dim2=0
    g1[5] = (int)stride_elems;                     // tensor_dim0_stride[31:0]
    g1[6] = 0;                                     // stride0 hi | stride1 lo
    g1[7] = 0;                                     // stride1 hi
    i32x4 z4 = {0, 0, 0, 0};
#if defined(__clang_major__) && (__clang_major__ >= 23)
    i32x8 z8 = {0, 0, 0, 0, 0, 0, 0, 0};
    __builtin_amdgcn_tensor_load_to_lds(g0, g1, z4, z4, z8, 0);
#else
    __builtin_amdgcn_tensor_load_to_lds(g0, g1, z4, z4, 0);
#endif
}

__device__ __forceinline__ unsigned lds_addr_of(const void* p) {
    // generic pointers to LDS carry the offset in the low 32 bits (flat aperture)
    return (unsigned)(unsigned long long)p;
}

// ---------------------------------------------------------------------------
// elementwise f32 -> bf16
__global__ __launch_bounds__(256) void k_f32_to_bf16(const float* __restrict__ in,
                                                     bf16* __restrict__ out, int n) {
    int i = blockIdx.x * 256 + threadIdx.x;
    if (i < n) out[i] = (bf16)in[i];
}

// W [K,N] f32 (in,out layout) -> Wt [N,K] bf16 (K contiguous for WMMA B operand)
__global__ __launch_bounds__(256) void k_transpose_to_bf16(const float* __restrict__ in,
                                                           bf16* __restrict__ out,
                                                           int K, int N) {
    int i = blockIdx.x * 256 + threadIdx.x;
    if (i >= K * N) return;
    int n = i / K, k = i % K;
    out[i] = (bf16)in[(size_t)k * N + n];
}

// ---------------------------------------------------------------------------
// Generic WMMA GEMM, TDM-fed + LDS double buffered.
// Out[M,N] = A[M,K] (bf16 row-major) @ Bt[N,K]^T + bias
// 256 threads = 8 waves; block tile 128x128, K-step 32; wave tile 32x64 (2x4 WMMAs).
template <bool RELU, bool OUT_BF16>
__global__ __launch_bounds__(256) void k_gemm(const bf16* __restrict__ A,
                                              const bf16* __restrict__ Bt,
                                              const float* __restrict__ bias,
                                              void* __restrict__ Out,
                                              int M, int N, int K) {
    __shared__ bf16 As[2][128][40];
    __shared__ bf16 Bs[2][128][40];

    const int ntiles = N / 128;
    const int tm = blockIdx.x / ntiles;
    const int tn = blockIdx.x % ntiles;
    const int t = threadIdx.x, lane = t & 31;
    const int wave = t >> 5;
    const int wm = wave >> 1, wn = wave & 1;     // 4x2 wave grid
    const int l16 = lane & 15, half = lane >> 4;
    const int waveu = wave_id_uniform();         // SGPR -> scalar branch

    const bf16* Abase = A  + (size_t)(tm * 128) * K;
    const bf16* Bbase = Bt + (size_t)(tn * 128) * K;
    const unsigned ldsA = lds_addr_of(&As[0][0][0]);
    const unsigned ldsB = lds_addr_of(&Bs[0][0][0]);
    const unsigned bufBytes = 128 * 40 * 2;      // 10240B per buffer

    if (waveu == 0) {
        tdm_load_tile_bf16(ldsA, Abase, K, 32, 3);
        tdm_load_tile_bf16(ldsB, Bbase, K, 32, 3);
    }

    v8f acc[2][4] = {};
    const int nk = K / 32;
    for (int i = 0; i < nk; ++i) {
        const int cur = i & 1;
        if (waveu == 0) {
            if (i + 1 < nk) {
                const unsigned nb = (unsigned)(cur ^ 1) * bufBytes;
                tdm_load_tile_bf16(ldsA + nb, Abase + (size_t)(i + 1) * 32, K, 32, 3);
                tdm_load_tile_bf16(ldsB + nb, Bbase + (size_t)(i + 1) * 32, K, 32, 3);
                __builtin_amdgcn_s_wait_tensorcnt(2);  // pair for `cur` is done
            } else {
                __builtin_amdgcn_s_wait_tensorcnt(0);
            }
        }
        __syncthreads();                                // LDS buf[cur] ready

        FragBf a[2], b[4];
        #pragma unroll
        for (int mt = 0; mt < 2; ++mt) {
            int r = wm * 32 + mt * 16 + l16;
            a[mt].h.lo = *(const v8bf*)&As[cur][r][half * 8];       // K 0-7 / 8-15
            a[mt].h.hi = *(const v8bf*)&As[cur][r][16 + half * 8];  // K 16-23 / 24-31
        }
        #pragma unroll
        for (int nt = 0; nt < 4; ++nt) {
            int c = wn * 64 + nt * 16 + l16;
            b[nt].h.lo = *(const v8bf*)&Bs[cur][c][half * 8];       // K 0-7 / 8-15
            b[nt].h.hi = *(const v8bf*)&Bs[cur][c][16 + half * 8];  // K 16-23 / 24-31
        }
        #pragma unroll
        for (int mt = 0; mt < 2; ++mt)
            #pragma unroll
            for (int nt = 0; nt < 4; ++nt)
                acc[mt][nt] = wmma_bf16(a[mt].v, b[nt].v, acc[mt][nt]);
        __syncthreads();                                // reads done before overwrite
    }

    #pragma unroll
    for (int mt = 0; mt < 2; ++mt)
        #pragma unroll
        for (int nt = 0; nt < 4; ++nt) {
            int gcol = tn * 128 + wn * 64 + nt * 16 + l16;
            float bv = bias ? bias[gcol] : 0.0f;
            #pragma unroll
            for (int j = 0; j < 8; ++j) {
                int grow = tm * 128 + wm * 32 + mt * 16 + half * 8 + j;
                float v = acc[mt][nt][j] + bv;
                if (RELU) v = v > 0.0f ? v : 0.0f;
                if (OUT_BF16) ((bf16*)Out)[(size_t)grow * N + gcol] = (bf16)v;
                else          ((float*)Out)[(size_t)grow * N + gcol] = v;
            }
        }
}

// ---------------------------------------------------------------------------
// Event detector tail: logit = hidden[row] . We2 + be2 ; sigmoid>0.5 <=> logit>0
__global__ __launch_bounds__(256) void k_event(const bf16* __restrict__ hidden,
                                               const float* __restrict__ We2,
                                               const float* __restrict__ be2,
                                               float* __restrict__ rowscale) {
    __shared__ float red[256];
    const int row = blockIdx.x, t = threadIdx.x;
    const bf16* hr = hidden + (size_t)row * (cH / 2);
    float s = 0.0f;
    for (int i = t; i < cH / 2; i += 256) s += (float)hr[i] * We2[i];
    red[t] = s;
    __syncthreads();
    for (int w = 128; w > 0; w >>= 1) {
        if (t < w) red[t] += red[t + w];
        __syncthreads();
    }
    if (t == 0) rowscale[row] = (red[0] + be2[0] > 0.0f) ? cScale : 0.0f;
}

// ---------------------------------------------------------------------------
// Per-head V transpose: VhT[b,h][d][k] = Vb[(b*S+k)*H + h*64 + d]
__global__ __launch_bounds__(256) void k_transpose_vh(const bf16* __restrict__ Vb,
                                                      bf16* __restrict__ VhT) {
    __shared__ bf16 tile[64][68];
    const int bh = blockIdx.y, b = bh / cNH, h = bh % cNH;
    const int k0 = blockIdx.x * 64;
    const int t = threadIdx.x;
    const int r = t >> 2, c0 = (t & 3) * 16;
    const bf16* src = Vb + (size_t)(b * cS + k0 + r) * cH + h * cHD + c0;
    #pragma unroll
    for (int i = 0; i < 16; ++i) tile[r][c0 + i] = src[i];
    __syncthreads();
    bf16* dst = VhT + ((size_t)bh * cHD + r) * cS + k0 + c0;
    #pragma unroll
    for (int i = 0; i < 16; ++i) dst[i] = tile[c0 + i][r];
}

// ---------------------------------------------------------------------------
// scores[b,h][q][k] = (Qh . Kh) * rowscale[q] * evw[h]   (raw, pre-softmax)
// Single-shot TDM stage: whole 128x64 A and B tiles (K = head_dim = 64),
// then two unrolled K-steps of 8 WMMAs. [128][72] pitch (128B rows + 16B pad).
__global__ __launch_bounds__(256) void k_scores(const bf16* __restrict__ Qb,
                                                const bf16* __restrict__ Kb,
                                                const float* __restrict__ rowscale,
                                                const float* __restrict__ evw,
                                                float* __restrict__ attn) {
    __shared__ bf16 As[128][72];
    __shared__ bf16 Bs[128][72];
    const int bh = blockIdx.y, b = bh / cNH, h = bh % cNH;
    const int ntiles = cS / 128;
    const int tm = blockIdx.x / ntiles, tn = blockIdx.x % ntiles;
    const int t = threadIdx.x, lane = t & 31;
    const int wave = t >> 5;
    const int wm = wave >> 1, wn = wave & 1;
    const int l16 = lane & 15, half = lane >> 4;
    const int waveu = wave_id_uniform();

    const bf16* Abase = Qb + (size_t)(b * cS + tm * 128) * cH + h * cHD;
    const bf16* Bbase = Kb + (size_t)(b * cS + tn * 128) * cH + h * cHD;
    float* outp = attn + (size_t)bh * cS * cS;

    if (waveu == 0) {
        tdm_load_tile_bf16(lds_addr_of(&As[0][0]), Abase, cH, 64, 4);
        tdm_load_tile_bf16(lds_addr_of(&Bs[0][0]), Bbase, cH, 64, 4);
        __builtin_amdgcn_s_wait_tensorcnt(0);
    }
    __syncthreads();

    v8f acc[2][4] = {};
    #pragma unroll
    for (int kk = 0; kk < 64; kk += 32) {
        FragBf a[2], bb[4];
        #pragma unroll
        for (int mt = 0; mt < 2; ++mt) {
            int r = wm * 32 + mt * 16 + l16;
            a[mt].h.lo = *(const v8bf*)&As[r][kk + half * 8];
            a[mt].h.hi = *(const v8bf*)&As[r][kk + 16 + half * 8];
        }
        #pragma unroll
        for (int nt = 0; nt < 4; ++nt) {
            int c = wn * 64 + nt * 16 + l16;
            bb[nt].h.lo = *(const v8bf*)&Bs[c][kk + half * 8];
            bb[nt].h.hi = *(const v8bf*)&Bs[c][kk + 16 + half * 8];
        }
        #pragma unroll
        for (int mt = 0; mt < 2; ++mt)
            #pragma unroll
            for (int nt = 0; nt < 4; ++nt)
                acc[mt][nt] = wmma_bf16(a[mt].v, bb[nt].v, acc[mt][nt]);
    }

    const float wh = evw[h];
    #pragma unroll
    for (int mt = 0; mt < 2; ++mt)
        #pragma unroll
        for (int nt = 0; nt < 4; ++nt) {
            int gcol = tn * 128 + wn * 64 + nt * 16 + l16;
            #pragma unroll
            for (int j = 0; j < 8; ++j) {
                int grow = tm * 128 + wm * 32 + mt * 16 + half * 8 + j;
                float rs = rowscale[b * cS + grow] * wh;
                outp[(size_t)grow * cS + gcol] = acc[mt][nt][j] * rs;
            }
        }
}

// ---------------------------------------------------------------------------
// In-place row softmax; whole row (2048 f32 = 8 KB) lives in LDS.
__global__ __launch_bounds__(256) void k_softmax(float* __restrict__ attn) {
    __shared__ float buf[cS];
    __shared__ float red[256];
    float* row = attn + (size_t)blockIdx.x * cS;
    const int t = threadIdx.x;

    float m = -3.4e38f;
    for (int i = t; i < cS; i += 256) { float v = row[i]; buf[i] = v; m = fmaxf(m, v); }
    red[t] = m;
    __syncthreads();
    for (int w = 128; w > 0; w >>= 1) { if (t < w) red[t] = fmaxf(red[t], red[t + w]); __syncthreads(); }
    const float rmax = red[0];
    __syncthreads();

    float s = 0.0f;
    for (int i = t; i < cS; i += 256) { float e = __expf(buf[i] - rmax); buf[i] = e; s += e; }
    red[t] = s;
    __syncthreads();
    for (int w = 128; w > 0; w >>= 1) { if (t < w) red[t] += red[t + w]; __syncthreads(); }
    const float inv = 1.0f / red[0];
    for (int i = t; i < cS; i += 256) row[i] = buf[i] * inv;
}

// ---------------------------------------------------------------------------
// ctx[b,s,h*64+d] = attn[b,h] @ Vh ; attn f32 -> bf16 on LDS fill (manual stage,
// TDM cannot convert). 8 waves; block tile 128(M) x 64(N), K-step 32.
__global__ __launch_bounds__(256) void k_ctx(const float* __restrict__ attn,
                                             const bf16* __restrict__ VhT,
                                             bf16* __restrict__ ctx) {
    __shared__ bf16 As[128][40];
    __shared__ bf16 Bs[64][40];
    const int bh = blockIdx.y, b = bh / cNH, h = bh % cNH;
    const int tm = blockIdx.x;
    const int t = threadIdx.x, lane = t & 31, wave = t >> 5;
    const int l16 = lane & 15, half = lane >> 4;

    const float* Arow = attn + (size_t)bh * cS * cS;
    const bf16* Bt = VhT + (size_t)bh * cHD * cS;   // [64, S]

    v8f acc[4] = {};
    const int lr = t >> 1, lc = (t & 1) * 16;       // A: 128 x 32 f32
    const int br = t >> 2, bc = (t & 3) * 8;        // B:  64 x 32 bf16

    for (int k0 = 0; k0 < cS; k0 += 32) {
        const float4* ap = (const float4*)(Arow + (size_t)(tm * 128 + lr) * cS + k0 + lc);
        #pragma unroll
        for (int i = 0; i < 4; ++i) {
            float4 v = ap[i];
            As[lr][lc + i * 4 + 0] = (bf16)v.x;
            As[lr][lc + i * 4 + 1] = (bf16)v.y;
            As[lr][lc + i * 4 + 2] = (bf16)v.z;
            As[lr][lc + i * 4 + 3] = (bf16)v.w;
        }
        *(v8bf*)&Bs[br][bc] = *(const v8bf*)(Bt + (size_t)br * cS + k0 + bc);
        __syncthreads();

        FragBf a;
        a.h.lo = *(const v8bf*)&As[wave * 16 + l16][half * 8];
        a.h.hi = *(const v8bf*)&As[wave * 16 + l16][16 + half * 8];
        #pragma unroll
        for (int nt = 0; nt < 4; ++nt) {
            FragBf bb;
            bb.h.lo = *(const v8bf*)&Bs[nt * 16 + l16][half * 8];
            bb.h.hi = *(const v8bf*)&Bs[nt * 16 + l16][16 + half * 8];
            acc[nt] = wmma_bf16(a.v, bb.v, acc[nt]);
        }
        __syncthreads();
    }

    #pragma unroll
    for (int nt = 0; nt < 4; ++nt)
        #pragma unroll
        for (int j = 0; j < 8; ++j) {
            int q = tm * 128 + wave * 16 + half * 8 + j;
            int d = nt * 16 + l16;
            ctx[(size_t)(b * cS + q) * cH + h * cHD + d] = (bf16)acc[nt][j];
        }
}

// ---------------------------------------------------------------------------
extern "C" void kernel_launch(void* const* d_in, const int* in_sizes, int n_in,
                              void* d_out, int out_size, void* d_ws, size_t ws_size,
                              hipStream_t stream) {
    const float* query = (const float*)d_in[0];
    const float* key_  = (const float*)d_in[1];
    const float* value = (const float*)d_in[2];
    const float* Wq  = (const float*)d_in[3];
    const float* bq  = (const float*)d_in[4];
    const float* Wk  = (const float*)d_in[5];
    const float* bk  = (const float*)d_in[6];
    const float* Wv  = (const float*)d_in[7];
    const float* bv  = (const float*)d_in[8];
    const float* Wo  = (const float*)d_in[9];
    const float* bo  = (const float*)d_in[10];
    const float* We1 = (const float*)d_in[11];
    const float* be1 = (const float*)d_in[12];
    const float* We2 = (const float*)d_in[13];
    const float* be2 = (const float*)d_in[14];
    const float* evw = (const float*)d_in[15];

    float* out  = (float*)d_out;                       // [B,S,H]
    float* attn = (float*)d_out + (size_t)cBS * cH;    // [B,NH,S,S]

    // workspace carve-up (~77 MB)
    char* ws = (char*)d_ws;
    size_t off = 0;
    auto take = [&](size_t bytes) -> char* {
        char* p = ws + off;
        off = (off + bytes + 255) & ~(size_t)255;
        return p;
    };
    bf16* Xq   = (bf16*)take((size_t)cBS * cH * 2);
    bf16* Xk   = (bf16*)take((size_t)cBS * cH * 2);
    bf16* Xv   = (bf16*)take((size_t)cBS * cH * 2);
    bf16* WqT  = (bf16*)take((size_t)cH * cH * 2);
    bf16* WkT  = (bf16*)take((size_t)cH * cH * 2);
    bf16* WvT  = (bf16*)take((size_t)cH * cH * 2);
    bf16* WoT  = (bf16*)take((size_t)cH * cH * 2);
    bf16* We1T = (bf16*)take((size_t)cH * (cH / 2) * 2);
    bf16* Qb   = (bf16*)take((size_t)cBS * cH * 2);
    bf16* Kb   = (bf16*)take((size_t)cBS * cH * 2);
    bf16* Vb   = (bf16*)take((size_t)cBS * cH * 2);
    bf16* hid  = (bf16*)take((size_t)cBS * (cH / 2) * 2);
    float* rsc = (float*)take((size_t)cBS * 4);
    bf16* VhT  = (bf16*)take((size_t)cB * cNH * cHD * cS * 2);
    bf16* ctx  = (bf16*)take((size_t)cBS * cH * 2);

    const int nAct = cBS * cH;   // 4,194,304

    // 1) conversions / weight transposes
    k_f32_to_bf16<<<nAct / 256, 256, 0, stream>>>(query, Xq, nAct);
    k_f32_to_bf16<<<nAct / 256, 256, 0, stream>>>(key_,  Xk, nAct);
    k_f32_to_bf16<<<nAct / 256, 256, 0, stream>>>(value, Xv, nAct);
    k_transpose_to_bf16<<<(cH * cH) / 256, 256, 0, stream>>>(Wq,  WqT,  cH, cH);
    k_transpose_to_bf16<<<(cH * cH) / 256, 256, 0, stream>>>(Wk,  WkT,  cH, cH);
    k_transpose_to_bf16<<<(cH * cH) / 256, 256, 0, stream>>>(Wv,  WvT,  cH, cH);
    k_transpose_to_bf16<<<(cH * cH) / 256, 256, 0, stream>>>(Wo,  WoT,  cH, cH);
    k_transpose_to_bf16<<<(cH * cH / 2) / 256, 256, 0, stream>>>(We1, We1T, cH, cH / 2);

    // 2) projections (WMMA, TDM-fed) and event MLP hidden layer (+ fused ReLU)
    const int gProj = (cBS / 128) * (cH / 128);          // 256 blocks
    k_gemm<false, true><<<gProj, 256, 0, stream>>>(Xq, WqT, bq, Qb, cBS, cH, cH);
    k_gemm<false, true><<<gProj, 256, 0, stream>>>(Xk, WkT, bk, Kb, cBS, cH, cH);
    k_gemm<false, true><<<gProj, 256, 0, stream>>>(Xv, WvT, bv, Vb, cBS, cH, cH);
    const int gHid = (cBS / 128) * ((cH / 2) / 128);     // 128 blocks
    k_gemm<true, true><<<gHid, 256, 0, stream>>>(Xq, We1T, be1, hid, cBS, cH / 2, cH);

    // 3) event threshold -> per-row score multiplier (scale * event)
    k_event<<<cBS, 256, 0, stream>>>(hid, We2, be2, rsc);

    // 4) per-head V transpose for the attn@V WMMA
    k_transpose_vh<<<dim3(cS / 64, cB * cNH), 256, 0, stream>>>(Vb, VhT);

    // 5) raw scores straight into the attn output region (WMMA, TDM-fed)
    k_scores<<<dim3((cS / 128) * (cS / 128), cB * cNH), 256, 0, stream>>>(Qb, Kb, rsc, evw, attn);

    // 6) in-place row softmax (attn is now final output #2)
    k_softmax<<<cB * cNH * cS, 256, 0, stream>>>(attn);

    // 7) ctx = attn @ V  (WMMA, attn converted to bf16 on the fly)
    k_ctx<<<dim3(cS / 128, cB * cNH), 256, 0, stream>>>(attn, VhT, ctx);

    // 8) out = ctx @ Wo + bo  (f32 output #1)
    k_gemm<false, false><<<gProj, 256, 0, stream>>>(ctx, WoT, bo, out, cBS, cH, cH);

    (void)in_sizes; (void)n_in; (void)out_size; (void)ws_size;
}
